// SPNet_36249523978292
// MI455X (gfx1250) — compile-verified, matrix operand order checked by reading, not run on hardware
//
#include <hip/hip_runtime.h>

typedef __attribute__((ext_vector_type(16))) _Float16 v16h;
typedef __attribute__((ext_vector_type(8)))  float    v8f;

#define WMMA_F16(a, b, c) \
  __builtin_amdgcn_wmma_f32_16x16x32_f16(false, (a), false, (b), (short)0, (c), false, false)

static constexpr int P = 3, C = 5, NF = 32;
static constexpr int N_HIT = 200000, N_SP = 100000, E = 300000;
static constexpr int FIN3D = P * NF;  // 96

// Per-lane K index for the 16-bit A/B 16x32 WMMA layout:
// lanes 0-15: K = {0..7, 16..23}; lanes 16-31: K = {8..15, 24..31}
__device__ __forceinline__ int kmap16(int e, int hi) {
  return (e < 8 ? e : e + 8) + hi * 8;
}

// ---------------------------------------------------------------- utilities

__global__ void zero_kernel(float* p, size_t n) {
  size_t i = (size_t)blockIdx.x * blockDim.x + threadIdx.x;
  size_t stride = (size_t)gridDim.x * blockDim.x;
  for (; i < n; i += stride) p[i] = 0.0f;
}

// One-shot f32 -> f16 weight conversion into pre-swizzled WMMA B-tile layout:
// out[mat][nt][ks][lane][e] = W[mat][nt*16 + (lane&15)][ks*32 + kmap16(e, lane>>4)]
// so each lane's B operand is one contiguous 32-byte v16h load.
__global__ void swizzle_weights_kernel(const float* __restrict__ W,
                                       _Float16* __restrict__ out,
                                       int nmat, int fout, int fin) {
  const int NT = fout >> 4, KS = fin >> 5;
  size_t total = (size_t)nmat * NT * KS * 512;
  size_t idx = (size_t)blockIdx.x * blockDim.x + threadIdx.x;
  if (idx >= total) return;
  int e    = (int)(idx & 15);
  int lane = (int)((idx >> 4) & 31);
  size_t t = idx >> 9;
  int ks  = (int)(t % KS);
  int nt  = (int)((t / KS) % NT);
  int mat = (int)(t / ((size_t)KS * NT));
  int n = nt * 16 + (lane & 15);
  int k = ks * 32 + kmap16(e, lane >> 4);
  out[idx] = (_Float16)W[((size_t)mat * fout + n) * fin + k];
}

// hit -> spacepoint scatter-add, building m_sp_in [N_SP][C][96]
__global__ void scatter_hits_kernel(const float* __restrict__ m,
                                    const int* __restrict__ hit_idx,
                                    const int* __restrict__ sp_idx,
                                    float* __restrict__ msp_in) {
  size_t idx = (size_t)blockIdx.x * blockDim.x + threadIdx.x;
  size_t total = (size_t)P * E * C * NF;
  if (idx >= total) return;
  int f = (int)(idx & 31);
  int c = (int)((idx >> 5) % C);
  size_t pe = idx / (C * NF);
  int e = (int)(pe % E);
  int p = (int)(pe / E);
  int h = hit_idx[(size_t)p * E + e];
  int s = sp_idx[(size_t)p * E + e];
  float v = m[(((size_t)p * N_HIT + h) * C + c) * NF + f];
  atomicAdd(&msp_in[((size_t)s * C + c) * FIN3D + p * NF + f], v);
}

// ------------------------------------------------------- node_net_3d (WMMA)
// one wave = 16 spacepoints x 1 class; 96->32 (tanh) -> 32->32 (tanh)
__global__ void node3d_kernel(const float* __restrict__ msp_in,
                              const _Float16* __restrict__ w1s, const float* __restrict__ b1,
                              const _Float16* __restrict__ w2s, const float* __restrict__ b2,
                              float* __restrict__ m_sp) {
  __shared__ _Float16 hbuf[8][16 * 32];
  const int wave = threadIdx.x >> 5;
  const int lane = threadIdx.x & 31;
  const int unit = blockIdx.x * 8 + wave;
  if (unit >= (N_SP / 16) * C) return;  // wave-uniform: EXEC stays all-1s
  const int c = unit % C;
  const int tile = (unit / C) * 16;
  const int hi = lane >> 4;
  const int mr = lane & 15;

  const float* Xrow = msp_in + ((size_t)(tile + mr) * C + c) * FIN3D;
  const _Float16* W1c = w1s + (size_t)c * 2 * 3 * 512;  // [nt][ks][512]

  v8f acc0 = {}, acc1 = {};
  for (int ks = 0; ks < 3; ++ks) {
    v16h a;
#pragma unroll
    for (int e = 0; e < 16; ++e) a[e] = (_Float16)Xrow[ks * 32 + kmap16(e, hi)];
    v16h bt0 = *(const v16h*)(W1c + (size_t)(0 * 3 + ks) * 512 + lane * 16);
    v16h bt1 = *(const v16h*)(W1c + (size_t)(1 * 3 + ks) * 512 + lane * 16);
    acc0 = WMMA_F16(a, bt0, acc0);
    acc1 = WMMA_F16(a, bt1, acc1);
  }
  const float bia0 = b1[c * 32 + mr], bia1 = b1[c * 32 + 16 + mr];
  _Float16* h = hbuf[wave];
#pragma unroll
  for (int r = 0; r < 8; ++r) {
    int M = r + hi * 8;
    h[M * 32 + mr]      = (_Float16)tanhf(acc0[r] + bia0);
    h[M * 32 + 16 + mr] = (_Float16)tanhf(acc1[r] + bia1);
  }
  // layer 2: K=32, N=32
  const _Float16* W2c = w2s + (size_t)c * 2 * 512;
  v16h a2;
#pragma unroll
  for (int e = 0; e < 16; ++e) a2[e] = h[mr * 32 + kmap16(e, hi)];
  v16h c0 = *(const v16h*)(W2c + (size_t)0 * 512 + lane * 16);
  v16h c1 = *(const v16h*)(W2c + (size_t)1 * 512 + lane * 16);
  v8f o0 = {}, o1 = {};
  o0 = WMMA_F16(a2, c0, o0);
  o1 = WMMA_F16(a2, c1, o1);
  const float b20 = b2[c * 32 + mr], b21 = b2[c * 32 + 16 + mr];
  float* dst = m_sp + ((size_t)tile * C + c) * 32;
#pragma unroll
  for (int r = 0; r < 8; ++r) {
    int M = r + hi * 8;
    dst[(size_t)M * C * 32 + mr]      = tanhf(o0[r] + b20);
    dst[(size_t)M * C * 32 + 16 + mr] = tanhf(o1[r] + b21);
  }
}

// --------------------------------------------- edge net + softmax + scatter
// one wave = 16 edges, all 5 classes (softmax couples classes)
__global__ void edge_kernel(int p,
                            const float* __restrict__ m,
                            const int* __restrict__ hit_idx,
                            const int* __restrict__ sp_idx,
                            const float* __restrict__ m_sp,
                            const _Float16* __restrict__ we1s, const float* __restrict__ be1,
                            const float* __restrict__ We2, const float* __restrict__ be2,
                            float* __restrict__ msg, float* __restrict__ cnt) {
  __shared__ int      sh_hit[8][16];
  __shared__ int      sh_sp[8][16];
  __shared__ float    sh_logit[8][C][16];
  __shared__ float    sh_a[8][16][C];
  __shared__ _Float16 sh_msp[8][16][C][32];  // gathered m_sp[sp] tile, reused for messages
  const int wave = threadIdx.x >> 5;
  const int lane = threadIdx.x & 31;
  const int unit = blockIdx.x * 8 + wave;
  if (unit >= E / 16) return;
  const int tile = unit * 16;
  const int hi = lane >> 4;
  const int mr = lane & 15;

  if (lane < 16) {
    sh_hit[wave][lane] = hit_idx[(size_t)p * E + tile + lane];
    sh_sp[wave][lane]  = sp_idx[(size_t)p * E + tile + lane];
  }
  const float* mp = m + (size_t)p * N_HIT * C * 32;
  const int myhit = sh_hit[wave][mr];
  const int mysp  = sh_sp[wave][mr];

  for (int c = 0; c < C; ++c) {
    const float* hrow = mp + ((size_t)myhit * C + c) * 32;
    const float* srow = m_sp + ((size_t)mysp * C + c) * 32;
    const _Float16* Wc = we1s + ((size_t)p * C + c) * 2 * 2 * 512;  // [nt][ks][512]
    // kstep 0: hit features; kstep 1: spacepoint features (stash to LDS)
    v16h aH, aS;
#pragma unroll
    for (int e = 0; e < 16; ++e) {
      int k = kmap16(e, hi);
      aH[e] = (_Float16)hrow[k];
      _Float16 v = (_Float16)srow[k];
      aS[e] = v;
      sh_msp[wave][mr][c][k] = v;
    }
    v16h b00 = *(const v16h*)(Wc + (size_t)(0 * 2 + 0) * 512 + lane * 16);
    v16h b01 = *(const v16h*)(Wc + (size_t)(0 * 2 + 1) * 512 + lane * 16);
    v16h b10 = *(const v16h*)(Wc + (size_t)(1 * 2 + 0) * 512 + lane * 16);
    v16h b11 = *(const v16h*)(Wc + (size_t)(1 * 2 + 1) * 512 + lane * 16);
    v8f a0 = {}, a1 = {};
    a0 = WMMA_F16(aH, b00, a0);
    a0 = WMMA_F16(aS, b01, a0);
    a1 = WMMA_F16(aH, b10, a1);
    a1 = WMMA_F16(aS, b11, a1);

    const float bb0 = be1[((size_t)p * C + c) * 32 + mr];
    const float bb1 = be1[((size_t)p * C + c) * 32 + 16 + mr];
    const float w20 = We2[((size_t)p * C + c) * 32 + mr];
    const float w21 = We2[((size_t)p * C + c) * 32 + 16 + mr];
#pragma unroll
    for (int r = 0; r < 8; ++r) {
      float v = tanhf(a0[r] + bb0) * w20 + tanhf(a1[r] + bb1) * w21;
      for (int off = 8; off >= 1; off >>= 1) v += __shfl_xor(v, off, 16);
      if (mr == 0) sh_logit[wave][c][r + hi * 8] = v + be2[(size_t)p * C + c];
    }
  }
  // class softmax per edge (lanes 0..15)
  if (lane < 16) {
    float mx = -1e30f;
    for (int c = 0; c < C; ++c) mx = fmaxf(mx, sh_logit[wave][c][lane]);
    float ev[C], s = 0.0f;
    for (int c = 0; c < C; ++c) { ev[c] = __expf(sh_logit[wave][c][lane] - mx); s += ev[c]; }
    float inv = 1.0f / s;
    for (int c = 0; c < C; ++c) sh_a[wave][lane][c] = ev[c] * inv;
    atomicAdd(&cnt[sh_hit[wave][lane]], 1.0f);
  }
  // messages: msg[hit] += a * m_sp[sp]  (m_sp tile served from LDS)
  for (int i = lane; i < 16 * C * 32; i += 32) {
    int f = i & 31;
    int rest = i >> 5;
    int c = rest % C;
    int e = rest / C;
    float val = sh_a[wave][e][c] * (float)sh_msp[wave][e][c][f];
    atomicAdd(&msg[((size_t)sh_hit[wave][e] * C + c) * 32 + f], val);
  }
}

// -------------------------------------------------------- node_net_2d (WMMA)
// one wave = 16 hits x 1 class; concat(m, msg/cnt) 64->32 (tanh) -> 32->32 (tanh)
__global__ void node2d_kernel(int p,
                              const float* __restrict__ m,
                              const float* __restrict__ msg,
                              const float* __restrict__ cnt,
                              const _Float16* __restrict__ wn1s, const float* __restrict__ bn1,
                              const _Float16* __restrict__ wn2s, const float* __restrict__ bn2,
                              float* __restrict__ out) {
  __shared__ _Float16 hbuf[8][16 * 32];
  const int wave = threadIdx.x >> 5;
  const int lane = threadIdx.x & 31;
  const int unit = blockIdx.x * 8 + wave;
  if (unit >= (N_HIT / 16) * C) return;
  const int c = unit % C;
  const int tile = (unit / C) * 16;
  const int hi = lane >> 4;
  const int mr = lane & 15;
  const int row = tile + mr;

  const float inv = 1.0f / fmaxf(cnt[row], 1.0f);
  const float* mrow = m + (((size_t)p * N_HIT + row) * C + c) * 32;
  const float* grow = msg + ((size_t)row * C + c) * 32;
  const _Float16* Wc = wn1s + ((size_t)p * C + c) * 2 * 2 * 512;  // [nt][ks][512]

  v16h aM, aG;
#pragma unroll
  for (int e = 0; e < 16; ++e) {
    int k = kmap16(e, hi);
    aM[e] = (_Float16)mrow[k];
    aG[e] = (_Float16)(grow[k] * inv);
  }
  v16h b00 = *(const v16h*)(Wc + (size_t)(0 * 2 + 0) * 512 + lane * 16);
  v16h b01 = *(const v16h*)(Wc + (size_t)(0 * 2 + 1) * 512 + lane * 16);
  v16h b10 = *(const v16h*)(Wc + (size_t)(1 * 2 + 0) * 512 + lane * 16);
  v16h b11 = *(const v16h*)(Wc + (size_t)(1 * 2 + 1) * 512 + lane * 16);
  v8f a0 = {}, a1 = {};
  a0 = WMMA_F16(aM, b00, a0);
  a0 = WMMA_F16(aG, b01, a0);
  a1 = WMMA_F16(aM, b10, a1);
  a1 = WMMA_F16(aG, b11, a1);

  const float bia0 = bn1[((size_t)p * C + c) * 32 + mr];
  const float bia1 = bn1[((size_t)p * C + c) * 32 + 16 + mr];
  _Float16* h = hbuf[wave];
#pragma unroll
  for (int r = 0; r < 8; ++r) {
    int M = r + hi * 8;
    h[M * 32 + mr]      = (_Float16)tanhf(a0[r] + bia0);
    h[M * 32 + 16 + mr] = (_Float16)tanhf(a1[r] + bia1);
  }
  const _Float16* W2c = wn2s + ((size_t)p * C + c) * 2 * 512;
  v16h a2;
#pragma unroll
  for (int e = 0; e < 16; ++e) a2[e] = h[mr * 32 + kmap16(e, hi)];
  v16h c0 = *(const v16h*)(W2c + (size_t)0 * 512 + lane * 16);
  v16h c1 = *(const v16h*)(W2c + (size_t)1 * 512 + lane * 16);
  v8f o0 = {}, o1 = {};
  o0 = WMMA_F16(a2, c0, o0);
  o1 = WMMA_F16(a2, c1, o1);
  const float b20 = bn2[((size_t)p * C + c) * 32 + mr];
  const float b21 = bn2[((size_t)p * C + c) * 32 + 16 + mr];
  float* dst = out + (((size_t)p * N_HIT + tile) * C + c) * 32;
#pragma unroll
  for (int r = 0; r < 8; ++r) {
    int M = r + hi * 8;
    dst[(size_t)M * C * 32 + mr]      = tanhf(o0[r] + b20);
    dst[(size_t)M * C * 32 + 16 + mr] = tanhf(o1[r] + b21);
  }
}

// ---------------------------------------------------------------- launcher

extern "C" void kernel_launch(void* const* d_in, const int* in_sizes, int n_in,
                              void* d_out, int out_size, void* d_ws, size_t ws_size,
                              hipStream_t stream) {
  const float* m       = (const float*)d_in[0];
  const int*   hit_idx = (const int*)d_in[1];
  const int*   sp_idx  = (const int*)d_in[2];
  const float* W1  = (const float*)d_in[3];
  const float* b1  = (const float*)d_in[4];
  const float* W2  = (const float*)d_in[5];
  const float* b2  = (const float*)d_in[6];
  const float* We1 = (const float*)d_in[7];
  const float* be1 = (const float*)d_in[8];
  const float* We2 = (const float*)d_in[9];
  const float* be2 = (const float*)d_in[10];
  const float* Wn1 = (const float*)d_in[11];
  const float* bn1 = (const float*)d_in[12];
  const float* Wn2 = (const float*)d_in[13];
  const float* bn2 = (const float*)d_in[14];
  float* out = (float*)d_out;

  // workspace layout (floats): msp_in[48M] | m_sp[16M] | cnt[200k] | f16 swizzled weights
  // msg[32M] overlays msp_in (dead after node3d; stream-ordered)
  float* msp_in = (float*)d_ws;
  float* m_sp   = msp_in + (size_t)N_SP * C * FIN3D;   // +48,000,000
  float* cnt    = m_sp + (size_t)N_SP * C * 32;        // +16,000,000
  _Float16* w1s  = (_Float16*)(cnt + N_HIT);
  _Float16* w2s  = w1s + (size_t)C * 2 * 3 * 512;          // 15360 halfs
  _Float16* we1s = w2s + (size_t)C * 2 * 1 * 512;          // 5120
  _Float16* wn1s = we1s + (size_t)P * C * 2 * 2 * 512;     // 30720
  _Float16* wn2s = wn1s + (size_t)P * C * 2 * 2 * 512;     // 30720
  float* msg = msp_in;

  // one-shot weight swizzle (tiny; runs every call for determinism)
  swizzle_weights_kernel<<<(C * 2 * 3 * 512 + 255) / 256, 256, 0, stream>>>(W1, w1s, C, 32, 96);
  swizzle_weights_kernel<<<(C * 2 * 1 * 512 + 255) / 256, 256, 0, stream>>>(W2, w2s, C, 32, 32);
  swizzle_weights_kernel<<<(P * C * 2 * 2 * 512 + 255) / 256, 256, 0, stream>>>(We1, we1s, P * C, 32, 64);
  swizzle_weights_kernel<<<(P * C * 2 * 2 * 512 + 255) / 256, 256, 0, stream>>>(Wn1, wn1s, P * C, 32, 64);
  swizzle_weights_kernel<<<(P * C * 2 * 1 * 512 + 255) / 256, 256, 0, stream>>>(Wn2, wn2s, P * C, 32, 32);

  const size_t n_mspin = (size_t)N_SP * C * FIN3D;
  zero_kernel<<<2048, 256, 0, stream>>>(msp_in, n_mspin);

  const size_t scatter_total = (size_t)P * E * C * NF;  // 144M
  scatter_hits_kernel<<<(unsigned)((scatter_total + 255) / 256), 256, 0, stream>>>(
      m, hit_idx, sp_idx, msp_in);

  const int units3d = (N_SP / 16) * C;  // 31250 waves
  node3d_kernel<<<(units3d + 7) / 8, 256, 0, stream>>>(msp_in, w1s, b1, w2s, b2, m_sp);

  const size_t n_msg = (size_t)N_HIT * C * 32;  // 32M
  for (int p = 0; p < P; ++p) {
    zero_kernel<<<2048, 256, 0, stream>>>(msg, n_msg);
    zero_kernel<<<256, 256, 0, stream>>>(cnt, (size_t)N_HIT);
    const int unitsE = E / 16;  // 18750 waves
    edge_kernel<<<(unitsE + 7) / 8, 256, 0, stream>>>(
        p, m, hit_idx, sp_idx, m_sp, we1s, be1, We2, be2, msg, cnt);
    const int units2d = (N_HIT / 16) * C;  // 62500 waves
    node2d_kernel<<<(units2d + 7) / 8, 256, 0, stream>>>(
        p, m, msg, cnt, wn1s, bn1, wn2s, bn2, out);
  }
}
